// NonLocalLayer_75771813036638
// MI455X (gfx1250) — matrix operand order repaired and checked by reading, not compiled, not run on hardware
//
#include <hip/hip_runtime.h>
#include <hip/hip_bf16.h>

#define B_   8
#define C_   128
#define MID_ 64
#define OUT_ 128
#define TN_  2048

typedef __attribute__((ext_vector_type(8)))  float  v8f;
typedef __attribute__((ext_vector_type(8)))  __bf16 v8bf;
typedef __attribute__((ext_vector_type(16))) __bf16 v16bf;

union V16 { v16bf v; v8bf h[2]; };

static __device__ inline v8f wmma_bf16(v16bf a, v16bf b, v8f c) {
  // D(16x16 f32) = A(16x32 bf16) * B(32x16 bf16) + C
  return __builtin_amdgcn_wmma_f32_16x16x32_bf16(false, a, false, b, (short)0, c, false, false);
}

// ---------------------------------------------------------------------------
// Weight f32 -> bf16 (all four weight matrices are 8192 elements each)
// ---------------------------------------------------------------------------
__global__ void cvt_w_kernel(const float* __restrict__ t, const float* __restrict__ p,
                             const float* __restrict__ g, const float* __restrict__ r,
                             __bf16* __restrict__ tb, __bf16* __restrict__ pb,
                             __bf16* __restrict__ gb, __bf16* __restrict__ rb) {
  int i = threadIdx.x + blockIdx.x * 256;
  if (i < MID_ * C_) {
    tb[i] = (__bf16)t[i];
    pb[i] = (__bf16)p[i];
    gb[i] = (__bf16)g[i];
    rb[i] = (__bf16)r[i];  // OUT_*MID_ == MID_*C_ == 8192
  }
}

// ---------------------------------------------------------------------------
// Kernel 1: WMMA projections theta/phi/g (+bias, ReLU), emit bf16
//   theta_t, phi_t : [B][TN][MID]   (row q contiguous over mid)
//   g_nt           : [B][MID][TN]   (row m contiguous over k)
// Per block: 64 q, 4 waves (16 q each). pc tile staged in LDS as bf16.
// ---------------------------------------------------------------------------
__global__ __launch_bounds__(128) void proj_kernel(
    const float* __restrict__ pc,
    const __bf16* __restrict__ twb, const float* __restrict__ tbias,
    const __bf16* __restrict__ pwb, const float* __restrict__ pbias,
    const __bf16* __restrict__ gwb, const float* __restrict__ gbias,
    __bf16* __restrict__ theta_t, __bf16* __restrict__ phi_t,
    __bf16* __restrict__ g_nt)
{
  const int QB = 64, XS = C_ + 8;        // row stride 136 bf16 = 68 dwords
  __shared__ __bf16 xs[QB * XS];         // ~17 KB
  int b  = blockIdx.x / (TN_ / QB);
  int q0 = (blockIdx.x % (TN_ / QB)) * QB;

  for (int i = threadIdx.x; i < C_ * QB; i += 128) {
    int c = i >> 6, qq = i & 63;
    xs[qq * XS + c] = (__bf16)pc[(size_t)b * C_ * TN_ + (size_t)c * TN_ + q0 + qq];
  }
  __syncthreads();

  int wave = threadIdx.x >> 5, lane = threadIdx.x & 31;
  int lo = lane & 15, hi = lane >> 4, ab = hi * 8;
  int qt = wave * 16;

  // A operand: 16 q rows, K = C (4 steps of 32)
  V16 a[4];
  #pragma unroll
  for (int s = 0; s < 4; ++s) {
    const __bf16* r = &xs[(qt + lo) * XS + s * 32];
    a[s].h[0] = *(const v8bf*)(r + hi * 16);
    a[s].h[1] = *(const v8bf*)(r + hi * 16 + 8);
  }

  const __bf16* wsel[3] = { twb, pwb, gwb };
  const float*  bsel[3] = { tbias, pbias, gbias };
  #pragma unroll
  for (int w = 0; w < 3; ++w) {
    #pragma unroll
    for (int mt = 0; mt < 4; ++mt) {
      const __bf16* wr = wsel[w] + (size_t)(mt * 16 + lo) * C_ + hi * 16;
      v8f acc = {};
      #pragma unroll
      for (int s = 0; s < 4; ++s) {
        v16bf bv = *(const v16bf*)(wr + s * 32);
        acc = wmma_bf16(a[s].v, bv, acc);
      }
      float bias = bsel[w][mt * 16 + lo];
      int m = mt * 16 + lo;
      #pragma unroll
      for (int r = 0; r < 8; ++r) {
        float v = fmaxf(acc[r] + bias, 0.f);
        int q = q0 + qt + r + ab;
        if (w == 0)      theta_t[((size_t)b * TN_ + q) * MID_ + m] = (__bf16)v;
        else if (w == 1) phi_t [((size_t)b * TN_ + q) * MID_ + m] = (__bf16)v;
        else             g_nt  [((size_t)b * MID_ + m) * TN_ + q] = (__bf16)v;
      }
    }
  }
}

// ---------------------------------------------------------------------------
// Kernel 2: per-key-column softmax stats over the query axis (online max/sum),
// scores recomputed with WMMA. Each wave owns 16 key columns.
// ---------------------------------------------------------------------------
__global__ __launch_bounds__(128) void stats_kernel(
    const __bf16* __restrict__ theta_t, const __bf16* __restrict__ phi_t,
    float* __restrict__ colmax, float* __restrict__ colinv)
{
  int wave = threadIdx.x >> 5, lane = threadIdx.x & 31;
  int lo = lane & 15, hi = lane >> 4, ab = hi * 8;
  int b  = blockIdx.x / (TN_ / 64);
  int k0 = (blockIdx.x % (TN_ / 64)) * 64 + wave * 16;
  int col = k0 + lo;

  // B operand (phi columns) hoisted: lane = column, K split by lane-half
  const __bf16* pr = phi_t + ((size_t)b * TN_ + col) * MID_;
  v16bf bv0 = *(const v16bf*)(pr + hi * 16);        // K(m) 0..31
  v16bf bv1 = *(const v16bf*)(pr + 32 + hi * 16);   // K(m) 32..63

  float mrun = -3.0e38f, srun = 0.f;
  const __bf16* tbase = theta_t + (size_t)b * TN_ * MID_;
  for (int q0 = 0; q0 < TN_; q0 += 16) {
    const __bf16* ar = tbase + (size_t)(q0 + lo) * MID_;
    V16 a0, a1;
    a0.h[0] = *(const v8bf*)(ar + hi * 16);
    a0.h[1] = *(const v8bf*)(ar + hi * 16 + 8);
    a1.h[0] = *(const v8bf*)(ar + 32 + hi * 16);
    a1.h[1] = *(const v8bf*)(ar + 32 + hi * 16 + 8);
    v8f s = {};
    s = wmma_bf16(a0.v, bv0, s);
    s = wmma_bf16(a1.v, bv1, s);

    float tm = s[0];
    #pragma unroll
    for (int r = 1; r < 8; ++r) tm = fmaxf(tm, s[r]);
    float nm = fmaxf(mrun, tm);
    float acc = 0.f;
    #pragma unroll
    for (int r = 0; r < 8; ++r) acc += __expf(s[r] - nm);
    srun = srun * __expf(mrun - nm) + acc;
    mrun = nm;
  }
  // combine the two lane-halves (M 0..7 with M 8..15)
  float mp = __shfl_xor(mrun, 16, 32);
  float sp = __shfl_xor(srun, 16, 32);
  float M  = fmaxf(mrun, mp);
  float S  = srun * __expf(mrun - M) + sp * __expf(mp - M);
  if (lane < 16) {
    colmax[b * TN_ + col] = M;
    colinv[b * TN_ + col] = 1.f / S;
  }
}

// ---------------------------------------------------------------------------
// Kernel 3: recompute scores -> corr -> y = corr*g -> refine -> permute + res.
// Each wave owns a 16-query tile; corr and y tiles round-trip through a
// wave-private LDS region (padded strides -> conflict-free ds_load_b128).
// ---------------------------------------------------------------------------
#define WSTR 40   // corr tile row stride (bf16): 20 dwords, conflict-free
#define YSTR 72   // y    tile row stride (bf16): 36 dwords, conflict-free
#define REGSZ (16 * YSTR)

__global__ __launch_bounds__(128) void attn_kernel(
    const __bf16* __restrict__ theta_t, const __bf16* __restrict__ phi_t,
    const __bf16* __restrict__ g_nt,
    const float* __restrict__ colmax, const float* __restrict__ colinv,
    const __bf16* __restrict__ rwb, const float* __restrict__ rbias,
    const float* __restrict__ pc, float* __restrict__ out)
{
  __shared__ __bf16 smem[4 * REGSZ];
  int wave = threadIdx.x >> 5, lane = threadIdx.x & 31;
  int lo = lane & 15, hi = lane >> 4, ab = hi * 8;
  int b  = blockIdx.x / (TN_ / 64);
  int q0 = (blockIdx.x % (TN_ / 64)) * 64 + wave * 16;
  __bf16* reg = &smem[wave * REGSZ];

  // Hoisted A operand: theta rows of this q-tile, K = mid (2 steps of 32)
  const __bf16* tr = theta_t + ((size_t)b * TN_ + (q0 + lo)) * MID_;
  V16 a0, a1;
  a0.h[0] = *(const v8bf*)(tr + hi * 16);
  a0.h[1] = *(const v8bf*)(tr + hi * 16 + 8);
  a1.h[0] = *(const v8bf*)(tr + 32 + hi * 16);
  a1.h[1] = *(const v8bf*)(tr + 32 + hi * 16 + 8);

  v8f zero = {};
  v8f acc[4];
  #pragma unroll
  for (int t = 0; t < 4; ++t) acc[t] = zero;

  for (int k0 = 0; k0 < TN_; k0 += 32) {
    // --- scores tile (16q x 32k) -> softmax weights -> LDS ---
    #pragma unroll
    for (int nt = 0; nt < 2; ++nt) {
      int kc = k0 + nt * 16 + lo;
      const __bf16* pr = phi_t + ((size_t)b * TN_ + kc) * MID_;
      v16bf bv0 = *(const v16bf*)(pr + hi * 16);
      v16bf bv1 = *(const v16bf*)(pr + 32 + hi * 16);
      v8f s = {};
      s = wmma_bf16(a0.v, bv0, s);
      s = wmma_bf16(a1.v, bv1, s);
      float cm = colmax[b * TN_ + kc];
      float ci = colinv[b * TN_ + kc];
      #pragma unroll
      for (int r = 0; r < 8; ++r) {
        float w = __expf(s[r] - cm) * ci;
        reg[(r + ab) * WSTR + nt * 16 + lo] = (__bf16)w;  // [q][k] tile
      }
    }
    asm volatile("s_wait_dscnt 0" ::: "memory");

    // --- reload corr tile as A operand, accumulate y += corr * g ---
    V16 aw;
    aw.h[0] = *(const v8bf*)&reg[lo * WSTR + hi * 16];
    aw.h[1] = *(const v8bf*)&reg[lo * WSTR + hi * 16 + 8];
    #pragma unroll
    for (int mt = 0; mt < 4; ++mt) {
      const __bf16* gr =
          g_nt + ((size_t)b * MID_ + mt * 16 + lo) * TN_ + k0 + hi * 16;
      v16bf bg = *(const v16bf*)gr;
      acc[mt] = wmma_bf16(aw.v, bg, acc[mt]);
    }
  }

  // --- y tile (16q x 64m) -> LDS bf16, then refine GEMM ---
  #pragma unroll
  for (int mt = 0; mt < 4; ++mt) {
    #pragma unroll
    for (int r = 0; r < 8; ++r)
      reg[(r + ab) * YSTR + mt * 16 + lo] = (__bf16)acc[mt][r];
  }
  asm volatile("s_wait_dscnt 0" ::: "memory");

  V16 ay0, ay1;
  ay0.h[0] = *(const v8bf*)&reg[lo * YSTR + hi * 16];
  ay0.h[1] = *(const v8bf*)&reg[lo * YSTR + hi * 16 + 8];
  ay1.h[0] = *(const v8bf*)&reg[lo * YSTR + 32 + hi * 16];
  ay1.h[1] = *(const v8bf*)&reg[lo * YSTR + 32 + hi * 16 + 8];

  #pragma unroll 2
  for (int ot = 0; ot < 8; ++ot) {
    const __bf16* rr0 = rwb + (size_t)(ot * 16 + lo) * MID_ + hi * 16;
    v16bf rb0 = *(const v16bf*)rr0;
    v16bf rb1 = *(const v16bf*)(rr0 + 32);
    v8f r = {};
    r = wmma_bf16(ay0.v, rb0, r);
    r = wmma_bf16(ay1.v, rb1, r);
    float bias = rbias[ot * 16 + lo];
    #pragma unroll
    for (int e = 0; e < 8; ++e) {
      float val = fmaxf(r[e] + bias, 0.f);
      int qp = q0 + e + ab;            // q' in refined[b][o'][q']
      int op = ot * 16 + lo;           // o'
      int flat = op * TN_ + qp;        // raw flat index of refined[b]
      int tt   = flat >> 16;           // reshape (T,N,OUT)
      int rem  = flat & 65535;
      int n    = rem >> 7;
      int o2   = rem & 127;
      size_t oidx = (((size_t)b * 128 + o2) * 4 + tt) * 512 + n;
      out[oidx] = val + pc[oidx];      // residual add
    }
  }
}

// ---------------------------------------------------------------------------
extern "C" void kernel_launch(void* const* d_in, const int* in_sizes, int n_in,
                              void* d_out, int out_size, void* d_ws, size_t ws_size,
                              hipStream_t stream) {
  (void)in_sizes; (void)n_in; (void)out_size; (void)ws_size;
  const float* pc = (const float*)d_in[0];
  const float* tw = (const float*)d_in[1];
  const float* tb = (const float*)d_in[2];
  const float* pw = (const float*)d_in[3];
  const float* pb = (const float*)d_in[4];
  const float* gw = (const float*)d_in[5];
  const float* gb = (const float*)d_in[6];
  const float* rw = (const float*)d_in[7];
  const float* rb = (const float*)d_in[8];
  float* out = (float*)d_out;

  char* ws = (char*)d_ws;
  size_t off = 0;
  __bf16* theta_t = (__bf16*)(ws + off); off += (size_t)B_ * TN_ * MID_ * 2;
  __bf16* phi_t   = (__bf16*)(ws + off); off += (size_t)B_ * TN_ * MID_ * 2;
  __bf16* g_nt    = (__bf16*)(ws + off); off += (size_t)B_ * MID_ * TN_ * 2;
  float*  colmax  = (float*) (ws + off); off += (size_t)B_ * TN_ * 4;
  float*  colinv  = (float*) (ws + off); off += (size_t)B_ * TN_ * 4;
  __bf16* twb     = (__bf16*)(ws + off); off += (size_t)MID_ * C_ * 2;
  __bf16* pwb     = (__bf16*)(ws + off); off += (size_t)MID_ * C_ * 2;
  __bf16* gwb     = (__bf16*)(ws + off); off += (size_t)MID_ * C_ * 2;
  __bf16* rwb     = (__bf16*)(ws + off); off += (size_t)OUT_ * MID_ * 2;

  cvt_w_kernel<<<32, 256, 0, stream>>>(tw, pw, gw, rw, twb, pwb, gwb, rwb);
  proj_kernel<<<B_ * (TN_ / 64), 128, 0, stream>>>(
      pc, twb, tb, pwb, pb, gwb, gb, theta_t, phi_t, g_nt);
  stats_kernel<<<B_ * (TN_ / 64), 128, 0, stream>>>(
      theta_t, phi_t, colmax, colinv);
  attn_kernel<<<B_ * (TN_ / 64), 128, 0, stream>>>(
      theta_t, phi_t, g_nt, colmax, colinv, rwb, rb, pc, out);
}
